// GATactilayer_27135603376743
// MI455X (gfx1250) — matrix operand order repaired and checked by reading, not compiled, not run on hardware
//
#include <hip/hip_runtime.h>
#include <hip/hip_bf16.h>

// ---------------------------------------------------------------------------
// GAT layer for MI455X (gfx1250, wave32, WMMA).
// Sizes fixed by the reference: N=8192, F_IN=1024, F_OUT=256.
// ---------------------------------------------------------------------------

#define GAT_N     8192
#define GAT_FIN   1024
#define GAT_FOUT  256
#define GAT_ALPHA 0.2f
#define GAT_NEGINF (-9.0e15f)

typedef __attribute__((ext_vector_type(16))) __bf16 v16bf;
typedef __attribute__((ext_vector_type(8)))  __bf16 v8bf;
typedef __attribute__((ext_vector_type(8)))  float  v8f;

// ---------------------------------------------------------------------------
// Fragment loaders for 16-bit WMMA using contiguous k order (k = 16h + s), same for A and B.
// A (16x32 bf16): lanes 0-15 -> M=lane, elems = K+0..15
//                 lanes 16-31 -> M=lane-16, elems = K+16..31
// B (32x16 bf16): lanes 0-15 -> N=lane,  elems = K+0..15
//                 lanes 16-31 -> N=lane-16, elems = K+16..31
// Both are contiguous per-lane reads from a row-major [rows, K] bf16 matrix,
// so A*B^T needs no transpose in memory (B-fragment lane = row of "Bt").
// ---------------------------------------------------------------------------
__device__ inline v16bf load_a_frag(const __bf16* A, int row, int K, int kk, int lane) {
    const int lo = (lane < 16) ? 0 : 16;
    const __bf16* p = A + (size_t)row * K + kk;
    v8bf x = *(const v8bf*)(p + lo);
    v8bf y = *(const v8bf*)(p + 8 + lo);
    return __builtin_shufflevector(x, y, 0,1,2,3,4,5,6,7,8,9,10,11,12,13,14,15);
}

__device__ inline v16bf load_b_frag(const __bf16* Bt, int col, int K, int kk, int lane) {
    const int base = (lane < 16) ? 0 : 16;
    const __bf16* p = Bt + (size_t)col * K + kk + base;
    return *(const v16bf*)p;
}

// ---------------------------------------------------------------------------
// Tiled bf16 WMMA GEMM: C[M,N] = A[M,K] * Bt[N,K]^T  (f32 accumulate).
// Workgroup = 8 waves (2 along M x 4 along N) -> 128x128 output tile.
// Wave tile = 64x32 via 4x2 grid of v_wmma_f32_16x16x32_bf16 accumulators.
// MODE 0: Wh epilogue   -> outf = Wh f32 [M,256], aux = Wh^T bf16 [256,M]
// MODE 1: comb epilogue -> outf = attention + A*A^T, zero diagonal
// MODE 2: ELU epilogue  -> outf = elu(A*Bt^T)
// ---------------------------------------------------------------------------
template <int MODE>
__global__ __launch_bounds__(256) void gat_gemm(
    const __bf16* __restrict__ A, const __bf16* __restrict__ Bt, int K,
    float* __restrict__ outf, const __bf16* aux, __bf16* auxw, int ldc, int M)
{
    const int lane = threadIdx.x & 31;
    const int wave = threadIdx.x >> 5;      // 0..7
    const int wm   = wave & 1;              // 0..1 along M
    const int wn   = wave >> 1;             // 0..3 along N
    const int i0   = blockIdx.y * 128 + wm * 64;
    const int j0   = blockIdx.x * 128 + wn * 32;
    const int nb   = lane & 15;

    v8f acc[4][2] = {};

    for (int kk = 0; kk < K; kk += 32) {
        v16bf af[4], bfr[2];
#pragma unroll
        for (int r = 0; r < 4; ++r)
            af[r] = load_a_frag(A, i0 + 16 * r + nb, K, kk, lane);
#pragma unroll
        for (int c = 0; c < 2; ++c)
            bfr[c] = load_b_frag(Bt, j0 + 16 * c + nb, K, kk, lane);

        if (kk + 64 <= K) {  // gfx1250 global_prefetch_b8 of next k-tile
            __builtin_prefetch(A  + (size_t)(i0 + nb) * K + kk + 32, 0, 1);
            __builtin_prefetch(Bt + (size_t)(j0 + nb) * K + kk + 32, 0, 1);
        }

#pragma unroll
        for (int r = 0; r < 4; ++r)
#pragma unroll
            for (int c = 0; c < 2; ++c)
                acc[r][c] = __builtin_amdgcn_wmma_f32_16x16x32_bf16(
                    false, af[r], false, bfr[c], (short)0, acc[r][c], false, false);
    }

    // Epilogue. C/D layout: elem t -> M = i0 + 16r + t + (lane<16 ? 0 : 8),
    //                              N = j0 + 16c + (lane & 15).
    const int me = (lane < 16) ? 0 : 8;
#pragma unroll
    for (int r = 0; r < 4; ++r) {
#pragma unroll
        for (int c = 0; c < 2; ++c) {
            const int j = j0 + 16 * c + nb;
#pragma unroll
            for (int t = 0; t < 8; ++t) {
                const int i = i0 + 16 * r + me + t;
                float v = acc[r][c][t];
                if (MODE == 0) {
                    outf[(size_t)i * ldc + j] = v;
                    auxw[(size_t)j * M + i]   = (__bf16)v;   // Wh^T bf16
                } else if (MODE == 1) {
                    v += (float)aux[(size_t)i * K + j];      // + attention
                    if (i == j) v = 0.0f;                    // zero diagonal
                    outf[(size_t)i * ldc + j] = v;
                } else {
                    outf[(size_t)i * ldc + j] = (v > 0.0f) ? v : (expf(v) - 1.0f);
                }
            }
        }
    }
}

// ---------------------------------------------------------------------------
// f32 -> bf16 conversions
// ---------------------------------------------------------------------------
__global__ void cvt_bf16(const float* __restrict__ src, __bf16* __restrict__ dst, int n) {
    for (int i = blockIdx.x * blockDim.x + threadIdx.x; i < n; i += gridDim.x * blockDim.x)
        dst[i] = (__bf16)src[i];
}

// W [1024,256] row-major -> Wt bf16 [256,1024] (transposed)
__global__ void cvt_W_t(const float* __restrict__ W, __bf16* __restrict__ Wt) {
    int idx = blockIdx.x * blockDim.x + threadIdx.x;
    if (idx < GAT_FIN * GAT_FOUT) {
        int k = idx >> 8;         // 0..1023
        int n = idx & 255;        // 0..255
        Wt[(size_t)n * GAT_FIN + k] = (__bf16)W[idx];
    }
}

// ---------------------------------------------------------------------------
// Wh1[m] = Wh[m,:] . a[0:256],  Wh2[m] = Wh[m,:] . a[256:512]
// ---------------------------------------------------------------------------
__global__ __launch_bounds__(256) void gat_wh12(
    const float* __restrict__ Wh, const float* __restrict__ a,
    float* __restrict__ Wh1, float* __restrict__ Wh2)
{
    __shared__ float r1[256], r2[256];
    const int row = blockIdx.x, t = threadIdx.x;
    const float p = Wh[(size_t)row * GAT_FOUT + t];
    r1[t] = p * a[t];
    r2[t] = p * a[GAT_FOUT + t];
    __syncthreads();
    for (int s = 128; s > 0; s >>= 1) {
        if (t < s) { r1[t] += r1[t + s]; r2[t] += r2[t + s]; }
        __syncthreads();
    }
    if (t == 0) { Wh1[row] = r1[0]; Wh2[row] = r2[0]; }
}

// ---------------------------------------------------------------------------
// Per-row masked softmax -> bf16 attention; fused degree count s2.
// One 256-thread workgroup per row; row cached in 32 KB LDS (320 KB/WGP).
// ---------------------------------------------------------------------------
__global__ __launch_bounds__(256) void gat_softmax(
    const float* __restrict__ adj, const float* __restrict__ Wh1,
    const float* __restrict__ Wh2, __bf16* __restrict__ attn, int* __restrict__ s2)
{
    __shared__ float z[GAT_N];
    __shared__ float rf[256];
    __shared__ int   ri[256];
    const int row = blockIdx.x, t = threadIdx.x;
    const float* arow = adj + (size_t)row * GAT_N;
    const float w1 = Wh1[row];

    float lmax = -3.4e38f;
    int lcnt = 0;
    for (int j = t; j < GAT_N; j += 256) {
        float e = w1 + Wh2[j];
        e = (e > 0.0f) ? e : GAT_ALPHA * e;            // LeakyReLU
        const float aj = arow[j];
        const float zz = (aj > 0.0f) ? e : GAT_NEGINF; // mask (finite, like ref)
        z[j] = zz;
        lmax = fmaxf(lmax, zz);
        lcnt += (aj != 0.0f && j != row) ? 1 : 0;      // s2 degree count
    }
    rf[t] = lmax; ri[t] = lcnt;
    __syncthreads();
    for (int s = 128; s > 0; s >>= 1) {
        if (t < s) { rf[t] = fmaxf(rf[t], rf[t + s]); ri[t] += ri[t + s]; }
        __syncthreads();
    }
    const float m = rf[0];
    if (t == 0) s2[row] = ri[0];
    __syncthreads();

    float lsum = 0.0f;
    for (int j = t; j < GAT_N; j += 256) {
        const float ez = expf(z[j] - m);
        z[j] = ez;
        lsum += ez;
    }
    rf[t] = lsum;
    __syncthreads();
    for (int s = 128; s > 0; s >>= 1) {
        if (t < s) rf[t] += rf[t + s];
        __syncthreads();
    }
    const float inv = 1.0f / rf[0];
    for (int j = t; j < GAT_N; j += 256)
        attn[(size_t)row * GAT_N + j] = (__bf16)(z[j] * inv);
}

// ---------------------------------------------------------------------------
// Per-row exact k-th-largest threshold (k = s2[row]) via 32-step MSB radix
// select on float bit patterns (monotonic: comb >= 0), then rewrite the row
// in place as the 0/1 adj_resize mask.
// ---------------------------------------------------------------------------
__global__ __launch_bounds__(256) void gat_prune(
    float* __restrict__ out_adj, const int* __restrict__ s2)
{
    __shared__ float z[GAT_N];
    __shared__ unsigned cnt_s;
    const int row = blockIdx.x, t = threadIdx.x;
    float* crow = out_adj + (size_t)row * GAT_N;
    const int k = s2[row];

    for (int j = t; j < GAT_N; j += 256) z[j] = crow[j];
    __syncthreads();

    if (k <= 0) {  // uniform per block: rows with no off-diagonal edges -> all zero
        for (int j = t; j < GAT_N; j += 256) crow[j] = 0.0f;
        return;
    }

    unsigned prefix = 0;
    for (int b = 31; b >= 0; --b) {
        const unsigned cand = prefix | (1u << b);
        if (t == 0) cnt_s = 0;
        __syncthreads();
        unsigned lc = 0;
        for (int j = t; j < GAT_N; j += 256)
            lc += (__float_as_uint(z[j]) >= cand) ? 1u : 0u;
        atomicAdd(&cnt_s, lc);
        __syncthreads();
        if (cnt_s >= (unsigned)k) prefix = cand;   // largest thr with count>=k
        __syncthreads();
    }
    const float thr = __uint_as_float(prefix);

    for (int j = t; j < GAT_N; j += 256) {
        const float v = z[j];
        crow[j] = (v >= thr && v > 0.0f) ? 1.0f : 0.0f;
    }
}

// ---------------------------------------------------------------------------
// Launch: inputs {h, adj, W, a}; outputs [elu(h_prime) | adj_resize].
// Workspace (~165 MB): attn bf16 fits the 192 MB L2 so the 2*8192^3 GEMM
// streams operands from L2, compute-bound on v_wmma_f32_16x16x32_bf16.
// comb is staged in the adj_resize output region and pruned in place.
// ---------------------------------------------------------------------------
extern "C" void kernel_launch(void* const* d_in, const int* in_sizes, int n_in,
                              void* d_out, int out_size, void* d_ws, size_t ws_size,
                              hipStream_t stream) {
    const float* h   = (const float*)d_in[0];
    const float* adj = (const float*)d_in[1];
    const float* W   = (const float*)d_in[2];
    const float* a   = (const float*)d_in[3];

    float* out_hp  = (float*)d_out;                       // [8192,256]
    float* out_adj = out_hp + (size_t)GAT_N * GAT_FOUT;   // [8192,8192]

    char* p = (char*)d_ws;
    __bf16* attn_bf = (__bf16*)p; p += (size_t)GAT_N * GAT_N * 2;        // 128 MB
    __bf16* h_bf    = (__bf16*)p; p += (size_t)GAT_N * GAT_FIN * 2;      // 16 MB
    __bf16* Wt_bf   = (__bf16*)p; p += (size_t)GAT_FOUT * GAT_FIN * 2;   // 0.5 MB
    float*  Wh      = (float*)p;  p += (size_t)GAT_N * GAT_FOUT * 4;     // 8 MB
    __bf16* Wht_bf  = (__bf16*)p; p += (size_t)GAT_FOUT * GAT_N * 2;     // 4 MB
    float*  Wh1     = (float*)p;  p += GAT_N * 4;
    float*  Wh2     = (float*)p;  p += GAT_N * 4;
    int*    s2      = (int*)p;    p += GAT_N * 4;

    // 1) bf16 conversions
    cvt_bf16<<<4096, 256, 0, stream>>>(h, h_bf, GAT_N * GAT_FIN);
    cvt_W_t<<<(GAT_FIN * GAT_FOUT) / 256, 256, 0, stream>>>(W, Wt_bf);

    // 2) Wh = h @ W  (also emits Wh^T bf16 for the h_prime GEMM)
    dim3 g0(GAT_FOUT / 128, GAT_N / 128);
    gat_gemm<0><<<g0, 256, 0, stream>>>(h_bf, Wt_bf, GAT_FIN, Wh, nullptr, Wht_bf,
                                        GAT_FOUT, GAT_N);

    // 3) Wh1 / Wh2 projections
    gat_wh12<<<GAT_N, 256, 0, stream>>>(Wh, a, Wh1, Wh2);

    // 4) masked softmax rows -> bf16 attention, fused s2 degree count
    gat_softmax<<<GAT_N, 256, 0, stream>>>(adj, Wh1, Wh2, attn_bf, s2);

    // 5) comb = attention + attention @ attention^T (diag zeroed) -> out_adj
    dim3 g1(GAT_N / 128, GAT_N / 128);
    gat_gemm<1><<<g1, 256, 0, stream>>>(attn_bf, attn_bf, GAT_N, out_adj, attn_bf,
                                        nullptr, GAT_N, GAT_N);

    // 6) h_prime = elu(attention @ Wh) -> out_hp
    dim3 g2(GAT_FOUT / 128, GAT_N / 128);
    gat_gemm<2><<<g2, 256, 0, stream>>>(attn_bf, Wht_bf, GAT_N, out_hp, nullptr,
                                        nullptr, GAT_FOUT, GAT_N);

    // 7) per-row k-th-largest threshold + binarize, in place on out_adj
    gat_prune<<<GAT_N, 256, 0, stream>>>(out_adj, s2);
}
